// DecoderLayer_47459388621582
// MI455X (gfx1250) — matrix operand order repaired and checked by reading, not compiled, not run on hardware
//
#include <hip/hip_runtime.h>
#include <hip/hip_bf16.h>
#include <math.h>

typedef __attribute__((ext_vector_type(16))) __bf16 v16bf;
typedef __attribute__((ext_vector_type(8)))  __bf16 v8bf;
typedef __attribute__((ext_vector_type(4)))  __bf16 v4bf;
typedef __attribute__((ext_vector_type(8)))  float  v8f;
typedef __attribute__((ext_vector_type(4)))  float  v4f;

#define D_MODEL 1024
#define N_HEADS 16
#define DK      64
#define D_FF    4096
#define BATCH   4
#define SEQ     2048
#define MROWS   (BATCH * SEQ)
#define LN_EPS  1e-6f

#define KC      64            // K-chunk staged in LDS
#define BROW    80            // LDS row stride (bf16): 160B, banks staggered, 32B aligned

// ---------------------------------------------------------------------------
// fp32 -> bf16 conversion, 4 elements per thread (all arrays are /4)
// ---------------------------------------------------------------------------
__global__ __launch_bounds__(256)
void cvt_f32_bf16_kernel(const float* __restrict__ in, __bf16* __restrict__ out, int n4) {
    int i = blockIdx.x * blockDim.x + threadIdx.x;
    const int stride = gridDim.x * blockDim.x;
    const v4f* in4  = (const v4f*)in;
    v4bf*      out4 = (v4bf*)out;
    for (; i < n4; i += stride) {
        v4f f = in4[i];
        v4bf o;
        #pragma unroll
        for (int j = 0; j < 4; ++j) o[j] = (__bf16)f[j];
        out4[i] = o;
    }
}

// ---------------------------------------------------------------------------
// WMMA bf16 GEMM:  C[M,N] = alpha * (A[M,K] x B[N,K]^T) + bias
//   B is [N,K] row-major. B tile (64N x 64K) is staged in LDS once per block
//   and shared by all 8 waves (8x global-traffic reduction vs direct loads).
//   causal : C[m][n] = -1e9 where n > m   |   relu : max(0,.)
// Wave tile = 16(M) x 64(N): 1 A fragment, 4 B fragments from LDS,
// 4 independent accumulators -> no WMMA->WMMA RAW hazard chain.
// Block = 256 threads = 8 waves stacked in M -> 128(M) x 64(N).
// Grids tile exactly; no early-out (barriers inside the K loop).
// ---------------------------------------------------------------------------
__global__ __launch_bounds__(256)
void gemm_wmma_bf16_kernel(const __bf16* __restrict__ A, int lda,
                           const __bf16* __restrict__ Bm, int ldb,
                           const float* __restrict__ bias,
                           float alpha, int causal, int relu,
                           float* __restrict__ Cf, __bf16* __restrict__ Cb, int ldc,
                           int M, int N, int K) {
    __shared__ __attribute__((aligned(32))) __bf16 Bs[64 * BROW];

    const int tid   = threadIdx.x;
    const int lane  = tid & 31;
    const int wave  = tid >> 5;
    const int tileM = blockIdx.y * 128 + wave * 16;
    const int tileN = blockIdx.x * 64;
    const int half  = lane >> 4;               // 0 / 1
    const int l16   = lane & 15;

    v8f acc[4] = {};

    // A fragment source (contiguous k = 16*half + s, matching B), straight from global
    const __bf16* ap = A + (size_t)(tileM + l16) * lda + half * 16;

    // B staging: thread t loads 16 contiguous bf16 of row (tileN + t/4)
    const int br = tid >> 2;                   // 0..63 (N row within tile)
    const int bc = (tid & 3) * 16;             // 0,16,32,48 (K col group)
    const __bf16* bsrc = Bm + (size_t)(tileN + br) * ldb + bc;
    __bf16*       bdst = &Bs[br * BROW + bc];

    // per-lane LDS fragment base: row l16 (+nt*16), col half*16 (+kk)
    const __bf16* brd = &Bs[l16 * BROW + half * 16];

    for (int kc = 0; kc < K; kc += KC) {
        // stage 64x64 B tile: one 32B global load + one 32B LDS store per thread
        *(v16bf*)bdst = *(const v16bf*)(bsrc + kc);
        if (kc + KC < K) __builtin_prefetch(bsrc + kc + KC, 0, 0);
        __syncthreads();

        #pragma unroll
        for (int kk = 0; kk < KC; kk += 32) {
            const int k0 = kc + kk;
            v16bf a = *(const v16bf*)(ap + k0);
            v16bf b0 = *(const v16bf*)(brd +  0 * BROW + kk);
            v16bf b1 = *(const v16bf*)(brd + 16 * BROW + kk);
            v16bf b2 = *(const v16bf*)(brd + 32 * BROW + kk);
            v16bf b3 = *(const v16bf*)(brd + 48 * BROW + kk);
            acc[0] = __builtin_amdgcn_wmma_f32_16x16x32_bf16(false, a, false, b0, (short)0, acc[0], false, false);
            acc[1] = __builtin_amdgcn_wmma_f32_16x16x32_bf16(false, a, false, b1, (short)0, acc[1], false, false);
            acc[2] = __builtin_amdgcn_wmma_f32_16x16x32_bf16(false, a, false, b2, (short)0, acc[2], false, false);
            acc[3] = __builtin_amdgcn_wmma_f32_16x16x32_bf16(false, a, false, b3, (short)0, acc[3], false, false);
        }
        __syncthreads();
    }

    // epilogue: C/D layout -> m = r + 8*(lane>>4), n = lane&15 (per subtile)
    #pragma unroll
    for (int nt = 0; nt < 4; ++nt) {
        const int gn = tileN + nt * 16 + l16;
        const float bval = bias ? bias[gn] : 0.0f;
        #pragma unroll
        for (int r = 0; r < 8; ++r) {
            const int gm = tileM + r + 8 * half;
            float v = alpha * acc[nt][r] + bval;
            if (causal && gn > gm) v = -1e9f;
            if (relu && v < 0.0f)  v = 0.0f;
            const size_t idx = (size_t)gm * ldc + gn;
            if (Cf) Cf[idx] = v;
            if (Cb) Cb[idx] = (__bf16)v;
        }
    }
}

// ---------------------------------------------------------------------------
// Per-(b,h) transpose of V: [B,S,D] slice (ld=D_MODEL) -> Vt[B*H][DK][SEQ].
// LDS-tiled 64x64, padded to kill bank conflicts. Coalesced both directions.
// ---------------------------------------------------------------------------
__global__ __launch_bounds__(256)
void transpose_v_kernel(const __bf16* __restrict__ Vb, __bf16* __restrict__ Vt) {
    __shared__ __bf16 tile[64][65];
    const int bh = blockIdx.y;            // 0..B*H-1
    const int s0 = blockIdx.x * 64;
    const int b  = bh >> 4, h = bh & 15;
    const __bf16* src = Vb + (size_t)b * SEQ * D_MODEL + (size_t)h * DK; // [s][d]
    __bf16*       dst = Vt + (size_t)bh * DK * SEQ;                      // [d][s]

    const int r  = threadIdx.x >> 2;          // 0..63
    const int c0 = (threadIdx.x & 3) * 16;    // 0,16,32,48
    #pragma unroll
    for (int j = 0; j < 16; ++j)
        tile[r][c0 + j] = src[(size_t)(s0 + r) * D_MODEL + c0 + j];
    __syncthreads();
    #pragma unroll
    for (int j = 0; j < 16; ++j)
        dst[(size_t)r * SEQ + s0 + c0 + j] = tile[c0 + j][r];
}

// ---------------------------------------------------------------------------
// Row softmax (fp32 in, bf16 out). One block per row, ncols = SEQ = 2048.
// ---------------------------------------------------------------------------
__global__ __launch_bounds__(256)
void softmax_row_kernel(const float* __restrict__ S, __bf16* __restrict__ P, int ncols) {
    __shared__ float red[256];
    const int row = blockIdx.x;
    const int t   = threadIdx.x;
    const float* s = S + (size_t)row * ncols;
    __bf16*      p = P + (size_t)row * ncols;

    float v[8];
    float mx = -3.4e38f;
    #pragma unroll
    for (int j = 0; j < 8; ++j) { v[j] = s[t + j * 256]; mx = fmaxf(mx, v[j]); }
    red[t] = mx; __syncthreads();
    for (int off = 128; off > 0; off >>= 1) {
        if (t < off) red[t] = fmaxf(red[t], red[t + off]);
        __syncthreads();
    }
    mx = red[0]; __syncthreads();

    float sum = 0.0f;
    #pragma unroll
    for (int j = 0; j < 8; ++j) { v[j] = __expf(v[j] - mx); sum += v[j]; }
    red[t] = sum; __syncthreads();
    for (int off = 128; off > 0; off >>= 1) {
        if (t < off) red[t] += red[t + off];
        __syncthreads();
    }
    const float inv = 1.0f / red[0];
    #pragma unroll
    for (int j = 0; j < 8; ++j) p[t + j * 256] = (__bf16)(v[j] * inv);
}

// ---------------------------------------------------------------------------
// Residual add + LayerNorm (Bessel std, denom = std + eps). One block / row.
// ---------------------------------------------------------------------------
__global__ __launch_bounds__(256)
void resid_layernorm_kernel(const float* __restrict__ X, const float* __restrict__ Y,
                            const float* __restrict__ gamma, const float* __restrict__ beta,
                            float* __restrict__ Of, __bf16* __restrict__ Ob) {
    __shared__ float red[256];
    const int row = blockIdx.x;
    const int t   = threadIdx.x;
    const size_t base = (size_t)row * D_MODEL;

    float a[4];
    float s = 0.0f;
    #pragma unroll
    for (int j = 0; j < 4; ++j) {
        const int c = t + j * 256;
        a[j] = X[base + c] + Y[base + c];
        s += a[j];
    }
    red[t] = s; __syncthreads();
    for (int off = 128; off > 0; off >>= 1) {
        if (t < off) red[t] += red[t + off];
        __syncthreads();
    }
    const float mean = red[0] * (1.0f / D_MODEL);
    __syncthreads();

    float ss = 0.0f;
    #pragma unroll
    for (int j = 0; j < 4; ++j) { const float d = a[j] - mean; ss += d * d; }
    red[t] = ss; __syncthreads();
    for (int off = 128; off > 0; off >>= 1) {
        if (t < off) red[t] += red[t + off];
        __syncthreads();
    }
    const float stdv = sqrtf(red[0] / (float)(D_MODEL - 1));
    const float inv  = 1.0f / (stdv + LN_EPS);

    #pragma unroll
    for (int j = 0; j < 4; ++j) {
        const int c = t + j * 256;
        const float y = gamma[c] * (a[j] - mean) * inv + beta[c];
        if (Of) Of[base + c] = y;
        if (Ob) Ob[base + c] = (__bf16)y;
    }
}

// ---------------------------------------------------------------------------
extern "C" void kernel_launch(void* const* d_in, const int* in_sizes, int n_in,
                              void* d_out, int out_size, void* d_ws, size_t ws_size,
                              hipStream_t stream) {
    (void)in_sizes; (void)n_in; (void)out_size; (void)ws_size;
    const float* x      = (const float*)d_in[0];
    /* d_in[1] = mask (int) — causal mask applied analytically */
    const float* wq = (const float*)d_in[2];  const float* bq = (const float*)d_in[3];
    const float* wk = (const float*)d_in[4];  const float* bk = (const float*)d_in[5];
    const float* wv = (const float*)d_in[6];  const float* bv = (const float*)d_in[7];
    const float* wo = (const float*)d_in[8];  const float* bo = (const float*)d_in[9];
    const float* w1 = (const float*)d_in[10]; const float* b1 = (const float*)d_in[11];
    const float* w2 = (const float*)d_in[12]; const float* b2 = (const float*)d_in[13];
    const float* gamma1 = (const float*)d_in[14]; const float* beta1 = (const float*)d_in[15];
    const float* gamma2 = (const float*)d_in[16]; const float* beta2 = (const float*)d_in[17];
    float* out = (float*)d_out;

    // ---- workspace carve-out (256B aligned) ----
    char* ws = (char*)d_ws;
    size_t off = 0;
    auto carve = [&](size_t bytes) -> void* {
        void* p = ws + off;
        off += (bytes + 255) & ~(size_t)255;
        return p;
    };
    __bf16* xb    = (__bf16*)carve((size_t)MROWS * D_MODEL * 2);
    __bf16* wb    = (__bf16*)carve((size_t)D_FF  * D_MODEL * 2);   // reused per weight
    __bf16* Qb    = (__bf16*)carve((size_t)MROWS * D_MODEL * 2);
    __bf16* Kb    = (__bf16*)carve((size_t)MROWS * D_MODEL * 2);
    __bf16* Vb    = (__bf16*)carve((size_t)MROWS * D_MODEL * 2);
    __bf16* Vt    = (__bf16*)carve((size_t)MROWS * D_MODEL * 2);   // [B*H][DK][SEQ]
    float*  sc    = (float*) carve((size_t)SEQ * SEQ * 4);         // reused per (b,h)
    __bf16* Pb    = (__bf16*)carve((size_t)SEQ * SEQ * 2);         // reused per (b,h)
    __bf16* ctxb  = (__bf16*)carve((size_t)MROWS * D_MODEL * 2);
    float*  aout  = (float*) carve((size_t)MROWS * D_MODEL * 4);
    float*  hbuf  = (float*) carve((size_t)MROWS * D_MODEL * 4);
    __bf16* hb    = (__bf16*)carve((size_t)MROWS * D_MODEL * 2);
    __bf16* ff1b  = (__bf16*)carve((size_t)MROWS * D_FF * 2);
    float*  ff2   = (float*) carve((size_t)MROWS * D_MODEL * 4);

    const dim3 blk(256);
    auto cvt = [&](const float* src, __bf16* dst, int n) {
        cvt_f32_bf16_kernel<<<dim3(2048), blk, 0, stream>>>(src, dst, n / 4);
    };
    auto ggrid = [](int M, int N) { return dim3((unsigned)(N / 64), (unsigned)(M / 128)); };

    // ---- QKV projections: [8192,1024] x [1024,1024]^T, bf16 outputs ----
    cvt(x, xb, MROWS * D_MODEL);
    const float* wqkv[3] = { wq, wk, wv };
    const float* bqkv[3] = { bq, bk, bv };
    __bf16*      oqkv[3] = { Qb, Kb, Vb };
    for (int i = 0; i < 3; ++i) {
        cvt(wqkv[i], wb, D_MODEL * D_MODEL);
        gemm_wmma_bf16_kernel<<<ggrid(MROWS, D_MODEL), blk, 0, stream>>>(
            xb, D_MODEL, wb, D_MODEL, bqkv[i],
            1.0f, 0, 0, nullptr, oqkv[i], D_MODEL, MROWS, D_MODEL, D_MODEL);
    }

    // ---- transpose V per head: Vt[bh][d][s] ----
    transpose_v_kernel<<<dim3(SEQ / 64, BATCH * N_HEADS), blk, 0, stream>>>(Vb, Vt);

    // ---- attention per (batch, head) ----
    const float scale = 1.0f / 8.0f;  // 1/sqrt(64)
    for (int b = 0; b < BATCH; ++b) {
        for (int h = 0; h < N_HEADS; ++h) {
            const int    bh   = b * N_HEADS + h;
            const size_t qoff = (size_t)b * SEQ * D_MODEL + (size_t)h * DK;
            // scores = Q K^T / 8, causal -> fp32 [SEQ,SEQ]
            gemm_wmma_bf16_kernel<<<ggrid(SEQ, SEQ), blk, 0, stream>>>(
                Qb + qoff, D_MODEL, Kb + qoff, D_MODEL, nullptr,
                scale, /*causal=*/1, 0, sc, nullptr, SEQ, SEQ, SEQ, DK);
            // softmax rows -> bf16 P
            softmax_row_kernel<<<dim3(SEQ), blk, 0, stream>>>(sc, Pb, SEQ);
            // context = P x Vt^T : A=[SEQ,SEQ], B=Vt[bh] is [DK,SEQ] = [N,K]
            gemm_wmma_bf16_kernel<<<ggrid(SEQ, DK), blk, 0, stream>>>(
                Pb, SEQ, Vt + (size_t)bh * DK * SEQ, SEQ, nullptr,
                1.0f, 0, 0, nullptr, ctxb + qoff, D_MODEL, SEQ, DK, SEQ);
        }
    }

    // ---- output projection (fp32 out for residual) ----
    cvt(wo, wb, D_MODEL * D_MODEL);
    gemm_wmma_bf16_kernel<<<ggrid(MROWS, D_MODEL), blk, 0, stream>>>(
        ctxb, D_MODEL, wb, D_MODEL, bo,
        1.0f, 0, 0, aout, nullptr, D_MODEL, MROWS, D_MODEL, D_MODEL);

    // ---- residual + LN1 -> h (fp32) + hb (bf16) ----
    resid_layernorm_kernel<<<dim3(MROWS), blk, 0, stream>>>(
        x, aout, gamma1, beta1, hbuf, hb);

    // ---- FFN1: relu(h W1^T + b1) -> bf16 [8192,4096] ----
    cvt(w1, wb, D_FF * D_MODEL);
    gemm_wmma_bf16_kernel<<<ggrid(MROWS, D_FF), blk, 0, stream>>>(
        hb, D_MODEL, wb, D_MODEL, b1,
        1.0f, 0, /*relu=*/1, nullptr, ff1b, D_FF, MROWS, D_FF, D_MODEL);

    // ---- FFN2: ff1 W2^T + b2 -> fp32 [8192,1024] ----
    cvt(w2, wb, D_MODEL * D_FF);
    gemm_wmma_bf16_kernel<<<ggrid(MROWS, D_MODEL), blk, 0, stream>>>(
        ff1b, D_FF, wb, D_FF, b2,
        1.0f, 0, 0, ff2, nullptr, D_MODEL, MROWS, D_MODEL, D_FF);

    // ---- residual + LN2 -> final output ----
    resid_layernorm_kernel<<<dim3(MROWS), blk, 0, stream>>>(
        hbuf, ff2, gamma2, beta2, out, nullptr);
}
